// MarchingCubesCUDA_43696997269755
// MI455X (gfx1250) — compile-verified, matrix operand order checked by reading, not run on hardware
//
#include <hip/hip_runtime.h>
#include <stdint.h>

#define VDIM 128
#define CDIM 127
#define NCELL 2048383ull /* 127^3 */

typedef __attribute__((ext_vector_type(4)))  unsigned int u32x4;
typedef __attribute__((ext_vector_type(8)))  int          i32x8;
typedef __attribute__((ext_vector_type(4)))  int          i32x4;
typedef __attribute__((ext_vector_type(16))) _Float16     v16h;
typedef __attribute__((ext_vector_type(8)))  float        v8f;

// Marching-cubes triangle table (256 x 16, -1 padded), int8 to keep it to 4KB.
__constant__ __align__(16) signed char c_tri[256][16] = {
{-1,-1,-1,-1,-1,-1,-1,-1,-1,-1,-1,-1,-1,-1,-1,-1},
{0,8,3,-1,-1,-1,-1,-1,-1,-1,-1,-1,-1,-1,-1,-1},
{0,1,9,-1,-1,-1,-1,-1,-1,-1,-1,-1,-1,-1,-1,-1},
{1,8,3,9,8,1,-1,-1,-1,-1,-1,-1,-1,-1,-1,-1},
{1,2,10,-1,-1,-1,-1,-1,-1,-1,-1,-1,-1,-1,-1,-1},
{0,8,3,1,2,10,-1,-1,-1,-1,-1,-1,-1,-1,-1,-1},
{9,2,10,0,2,9,-1,-1,-1,-1,-1,-1,-1,-1,-1,-1},
{2,8,3,2,10,8,10,9,8,-1,-1,-1,-1,-1,-1,-1},
{3,11,2,-1,-1,-1,-1,-1,-1,-1,-1,-1,-1,-1,-1,-1},
{0,11,2,8,11,0,-1,-1,-1,-1,-1,-1,-1,-1,-1,-1},
{1,9,0,2,3,11,-1,-1,-1,-1,-1,-1,-1,-1,-1,-1},
{1,11,2,1,9,11,9,8,11,-1,-1,-1,-1,-1,-1,-1},
{3,10,1,11,10,3,-1,-1,-1,-1,-1,-1,-1,-1,-1,-1},
{0,10,1,0,8,10,8,11,10,-1,-1,-1,-1,-1,-1,-1},
{3,9,0,3,11,9,11,10,9,-1,-1,-1,-1,-1,-1,-1},
{9,8,10,10,8,11,-1,-1,-1,-1,-1,-1,-1,-1,-1,-1},
{4,7,8,-1,-1,-1,-1,-1,-1,-1,-1,-1,-1,-1,-1,-1},
{4,3,0,7,3,4,-1,-1,-1,-1,-1,-1,-1,-1,-1,-1},
{0,1,9,8,4,7,-1,-1,-1,-1,-1,-1,-1,-1,-1,-1},
{4,1,9,4,7,1,7,3,1,-1,-1,-1,-1,-1,-1,-1},
{1,2,10,8,4,7,-1,-1,-1,-1,-1,-1,-1,-1,-1,-1},
{3,4,7,3,0,4,1,2,10,-1,-1,-1,-1,-1,-1,-1},
{9,2,10,9,0,2,8,4,7,-1,-1,-1,-1,-1,-1,-1},
{2,10,9,2,9,7,2,7,3,7,9,4,-1,-1,-1,-1},
{8,4,7,3,11,2,-1,-1,-1,-1,-1,-1,-1,-1,-1,-1},
{11,4,7,11,2,4,2,0,4,-1,-1,-1,-1,-1,-1,-1},
{9,0,1,8,4,7,2,3,11,-1,-1,-1,-1,-1,-1,-1},
{4,7,11,9,4,11,9,11,2,9,2,1,-1,-1,-1,-1},
{3,10,1,3,11,10,7,8,4,-1,-1,-1,-1,-1,-1,-1},
{1,11,10,1,4,11,1,0,4,7,11,4,-1,-1,-1,-1},
{4,7,8,9,0,11,9,11,10,11,0,3,-1,-1,-1,-1},
{4,7,11,4,11,9,9,11,10,-1,-1,-1,-1,-1,-1,-1},
{9,5,4,-1,-1,-1,-1,-1,-1,-1,-1,-1,-1,-1,-1,-1},
{9,5,4,0,8,3,-1,-1,-1,-1,-1,-1,-1,-1,-1,-1},
{0,5,4,1,5,0,-1,-1,-1,-1,-1,-1,-1,-1,-1,-1},
{8,5,4,8,3,5,3,1,5,-1,-1,-1,-1,-1,-1,-1},
{1,2,10,9,5,4,-1,-1,-1,-1,-1,-1,-1,-1,-1,-1},
{3,0,8,1,2,10,4,9,5,-1,-1,-1,-1,-1,-1,-1},
{5,2,10,5,4,2,4,0,2,-1,-1,-1,-1,-1,-1,-1},
{2,10,5,3,2,5,3,5,4,3,4,8,-1,-1,-1,-1},
{9,5,4,2,3,11,-1,-1,-1,-1,-1,-1,-1,-1,-1,-1},
{0,11,2,0,8,11,4,9,5,-1,-1,-1,-1,-1,-1,-1},
{0,5,4,0,1,5,2,3,11,-1,-1,-1,-1,-1,-1,-1},
{2,1,5,2,5,8,2,8,11,4,8,5,-1,-1,-1,-1},
{10,3,11,10,1,3,9,5,4,-1,-1,-1,-1,-1,-1,-1},
{4,9,5,0,8,1,8,10,1,8,11,10,-1,-1,-1,-1},
{5,4,0,5,0,11,5,11,10,11,0,3,-1,-1,-1,-1},
{5,4,8,5,8,10,10,8,11,-1,-1,-1,-1,-1,-1,-1},
{9,7,8,5,7,9,-1,-1,-1,-1,-1,-1,-1,-1,-1,-1},
{9,3,0,9,5,3,5,7,3,-1,-1,-1,-1,-1,-1,-1},
{0,7,8,0,1,7,1,5,7,-1,-1,-1,-1,-1,-1,-1},
{1,5,3,3,5,7,-1,-1,-1,-1,-1,-1,-1,-1,-1,-1},
{9,7,8,9,5,7,10,1,2,-1,-1,-1,-1,-1,-1,-1},
{10,1,2,9,5,0,5,3,0,5,7,3,-1,-1,-1,-1},
{8,0,2,8,2,5,8,5,7,10,5,2,-1,-1,-1,-1},
{2,10,5,2,5,3,3,5,7,-1,-1,-1,-1,-1,-1,-1},
{7,9,5,7,8,9,3,11,2,-1,-1,-1,-1,-1,-1,-1},
{9,5,7,9,7,2,9,2,0,2,7,11,-1,-1,-1,-1},
{2,3,11,0,1,8,1,7,8,1,5,7,-1,-1,-1,-1},
{11,2,1,11,1,7,7,1,5,-1,-1,-1,-1,-1,-1,-1},
{9,5,8,8,5,7,10,1,3,10,3,11,-1,-1,-1,-1},
{5,7,0,5,0,9,7,11,0,1,0,10,11,10,0,-1},
{11,10,0,11,0,3,10,5,0,8,0,7,5,7,0,-1},
{11,10,5,7,11,5,-1,-1,-1,-1,-1,-1,-1,-1,-1,-1},
{10,6,5,-1,-1,-1,-1,-1,-1,-1,-1,-1,-1,-1,-1,-1},
{0,8,3,5,10,6,-1,-1,-1,-1,-1,-1,-1,-1,-1,-1},
{9,0,1,5,10,6,-1,-1,-1,-1,-1,-1,-1,-1,-1,-1},
{1,8,3,1,9,8,5,10,6,-1,-1,-1,-1,-1,-1,-1},
{1,6,5,2,6,1,-1,-1,-1,-1,-1,-1,-1,-1,-1,-1},
{1,6,5,1,2,6,3,0,8,-1,-1,-1,-1,-1,-1,-1},
{9,6,5,9,0,6,0,2,6,-1,-1,-1,-1,-1,-1,-1},
{5,9,8,5,8,2,5,2,6,3,2,8,-1,-1,-1,-1},
{2,3,11,10,6,5,-1,-1,-1,-1,-1,-1,-1,-1,-1,-1},
{11,0,8,11,2,0,10,6,5,-1,-1,-1,-1,-1,-1,-1},
{0,1,9,2,3,11,5,10,6,-1,-1,-1,-1,-1,-1,-1},
{5,10,6,1,9,2,9,11,2,9,8,11,-1,-1,-1,-1},
{6,3,11,6,5,3,5,1,3,-1,-1,-1,-1,-1,-1,-1},
{0,8,11,0,11,5,0,5,1,5,11,6,-1,-1,-1,-1},
{3,11,6,0,3,6,0,6,5,0,5,9,-1,-1,-1,-1},
{6,5,9,6,9,11,11,9,8,-1,-1,-1,-1,-1,-1,-1},
{5,10,6,4,7,8,-1,-1,-1,-1,-1,-1,-1,-1,-1,-1},
{4,3,0,4,7,3,6,5,10,-1,-1,-1,-1,-1,-1,-1},
{1,9,0,5,10,6,8,4,7,-1,-1,-1,-1,-1,-1,-1},
{10,6,5,1,9,7,1,7,3,7,9,4,-1,-1,-1,-1},
{6,1,2,6,5,1,4,7,8,-1,-1,-1,-1,-1,-1,-1},
{1,2,5,5,2,6,3,0,4,3,4,7,-1,-1,-1,-1},
{8,4,7,9,0,5,0,6,5,0,2,6,-1,-1,-1,-1},
{7,3,9,7,9,4,3,2,9,5,9,6,2,6,9,-1},
{3,11,2,7,8,4,10,6,5,-1,-1,-1,-1,-1,-1,-1},
{5,10,6,4,7,2,4,2,0,2,7,11,-1,-1,-1,-1},
{0,1,9,4,7,8,2,3,11,5,10,6,-1,-1,-1,-1},
{9,2,1,9,11,2,9,4,11,7,11,4,5,10,6,-1},
{8,4,7,3,11,5,3,5,1,5,11,6,-1,-1,-1,-1},
{5,1,11,5,11,6,1,0,11,7,11,4,0,4,11,-1},
{0,5,9,0,6,5,0,3,6,11,6,3,8,4,7,-1},
{6,5,9,6,9,11,4,7,9,7,11,9,-1,-1,-1,-1},
{10,4,9,6,4,10,-1,-1,-1,-1,-1,-1,-1,-1,-1,-1},
{4,10,6,4,9,10,0,8,3,-1,-1,-1,-1,-1,-1,-1},
{10,0,1,10,6,0,6,4,0,-1,-1,-1,-1,-1,-1,-1},
{8,3,1,8,1,6,8,6,4,6,1,10,-1,-1,-1,-1},
{1,4,9,1,2,4,2,6,4,-1,-1,-1,-1,-1,-1,-1},
{3,0,8,1,2,9,2,4,9,2,6,4,-1,-1,-1,-1},
{0,2,4,4,2,6,-1,-1,-1,-1,-1,-1,-1,-1,-1,-1},
{8,3,2,8,2,4,4,2,6,-1,-1,-1,-1,-1,-1,-1},
{10,4,9,10,6,4,11,2,3,-1,-1,-1,-1,-1,-1,-1},
{0,8,2,2,8,11,4,9,10,4,10,6,-1,-1,-1,-1},
{3,11,2,0,1,6,0,6,4,6,1,10,-1,-1,-1,-1},
{6,4,1,6,1,10,4,8,1,2,1,11,8,11,1,-1},
{9,6,4,9,3,6,9,1,3,11,6,3,-1,-1,-1,-1},
{8,11,1,8,1,0,11,6,1,9,1,4,6,4,1,-1},
{3,11,6,3,6,0,0,6,4,-1,-1,-1,-1,-1,-1,-1},
{6,4,8,11,6,8,-1,-1,-1,-1,-1,-1,-1,-1,-1,-1},
{7,10,6,7,8,10,8,9,10,-1,-1,-1,-1,-1,-1,-1},
{0,7,3,0,10,7,0,9,10,6,7,10,-1,-1,-1,-1},
{10,6,7,1,10,7,1,7,8,1,8,0,-1,-1,-1,-1},
{10,6,7,10,7,1,1,7,3,-1,-1,-1,-1,-1,-1,-1},
{1,2,6,1,6,8,1,8,9,8,6,7,-1,-1,-1,-1},
{2,6,9,2,9,1,6,7,9,0,9,3,7,3,9,-1},
{7,8,0,7,0,6,6,0,2,-1,-1,-1,-1,-1,-1,-1},
{7,3,2,6,7,2,-1,-1,-1,-1,-1,-1,-1,-1,-1,-1},
{2,3,11,10,6,8,10,8,9,8,6,7,-1,-1,-1,-1},
{2,0,7,2,7,11,0,9,7,6,7,10,9,10,7,-1},
{1,8,0,1,7,8,1,10,7,6,7,10,2,3,11,-1},
{11,2,1,11,1,7,10,6,1,6,7,1,-1,-1,-1,-1},
{8,9,6,8,6,7,9,1,6,11,6,3,1,3,6,-1},
{0,9,1,11,6,7,-1,-1,-1,-1,-1,-1,-1,-1,-1,-1},
{7,8,0,7,0,6,3,11,0,11,6,0,-1,-1,-1,-1},
{7,11,6,-1,-1,-1,-1,-1,-1,-1,-1,-1,-1,-1,-1,-1},
{7,6,11,-1,-1,-1,-1,-1,-1,-1,-1,-1,-1,-1,-1,-1},
{3,0,8,11,7,6,-1,-1,-1,-1,-1,-1,-1,-1,-1,-1},
{0,1,9,11,7,6,-1,-1,-1,-1,-1,-1,-1,-1,-1,-1},
{8,1,9,8,3,1,11,7,6,-1,-1,-1,-1,-1,-1,-1},
{10,1,2,6,11,7,-1,-1,-1,-1,-1,-1,-1,-1,-1,-1},
{1,2,10,3,0,8,6,11,7,-1,-1,-1,-1,-1,-1,-1},
{2,9,0,2,10,9,6,11,7,-1,-1,-1,-1,-1,-1,-1},
{6,11,7,2,10,3,10,8,3,10,9,8,-1,-1,-1,-1},
{7,2,3,6,2,7,-1,-1,-1,-1,-1,-1,-1,-1,-1,-1},
{7,0,8,7,6,0,6,2,0,-1,-1,-1,-1,-1,-1,-1},
{2,7,6,2,3,7,0,1,9,-1,-1,-1,-1,-1,-1,-1},
{1,6,2,1,8,6,1,9,8,8,7,6,-1,-1,-1,-1},
{10,7,6,10,1,7,1,3,7,-1,-1,-1,-1,-1,-1,-1},
{10,7,6,1,7,10,1,8,7,1,0,8,-1,-1,-1,-1},
{0,3,7,0,7,10,0,10,9,6,10,7,-1,-1,-1,-1},
{7,6,10,7,10,8,8,10,9,-1,-1,-1,-1,-1,-1,-1},
{6,8,4,11,8,6,-1,-1,-1,-1,-1,-1,-1,-1,-1,-1},
{3,6,11,3,0,6,0,4,6,-1,-1,-1,-1,-1,-1,-1},
{8,6,11,8,4,6,9,0,1,-1,-1,-1,-1,-1,-1,-1},
{9,4,6,9,6,3,9,3,1,11,3,6,-1,-1,-1,-1},
{6,8,4,6,11,8,2,10,1,-1,-1,-1,-1,-1,-1,-1},
{1,2,10,3,0,11,0,6,11,0,4,6,-1,-1,-1,-1},
{4,11,8,4,6,11,0,2,9,2,10,9,-1,-1,-1,-1},
{10,9,3,10,3,2,9,4,3,11,3,6,4,6,3,-1},
{8,2,3,8,4,2,4,6,2,-1,-1,-1,-1,-1,-1,-1},
{0,4,2,4,6,2,-1,-1,-1,-1,-1,-1,-1,-1,-1,-1},
{1,9,0,2,3,4,2,4,6,4,3,8,-1,-1,-1,-1},
{1,9,4,1,4,2,2,4,6,-1,-1,-1,-1,-1,-1,-1},
{8,1,3,8,6,1,8,4,6,6,10,1,-1,-1,-1,-1},
{10,1,0,10,0,6,6,0,4,-1,-1,-1,-1,-1,-1,-1},
{4,6,3,4,3,8,6,10,3,0,3,9,10,9,3,-1},
{10,9,4,6,10,4,-1,-1,-1,-1,-1,-1,-1,-1,-1,-1},
{4,9,5,7,6,11,-1,-1,-1,-1,-1,-1,-1,-1,-1,-1},
{0,8,3,4,9,5,11,7,6,-1,-1,-1,-1,-1,-1,-1},
{5,0,1,5,4,0,7,6,11,-1,-1,-1,-1,-1,-1,-1},
{11,7,6,8,3,4,3,5,4,3,1,5,-1,-1,-1,-1},
{9,5,4,10,1,2,7,6,11,-1,-1,-1,-1,-1,-1,-1},
{6,11,7,1,2,10,0,8,3,4,9,5,-1,-1,-1,-1},
{7,6,11,5,4,10,4,2,10,4,0,2,-1,-1,-1,-1},
{3,4,8,3,5,4,3,2,5,10,5,2,11,7,6,-1},
{7,2,3,7,6,2,5,4,9,-1,-1,-1,-1,-1,-1,-1},
{9,5,4,0,8,6,0,6,2,6,8,7,-1,-1,-1,-1},
{3,6,2,3,7,6,1,5,0,5,4,0,-1,-1,-1,-1},
{6,2,8,6,8,7,2,1,8,4,8,5,1,5,8,-1},
{9,5,4,10,1,6,1,7,6,1,3,7,-1,-1,-1,-1},
{1,6,10,1,7,6,1,0,7,8,7,0,9,5,4,-1},
{4,0,10,4,10,5,0,3,10,6,10,7,3,7,10,-1},
{7,6,10,7,10,8,5,4,10,4,8,10,-1,-1,-1,-1},
{6,9,5,6,11,9,11,8,9,-1,-1,-1,-1,-1,-1,-1},
{3,6,11,0,6,3,0,5,6,0,9,5,-1,-1,-1,-1},
{0,11,8,0,5,11,0,1,5,5,6,11,-1,-1,-1,-1},
{6,11,3,6,3,5,5,3,1,-1,-1,-1,-1,-1,-1,-1},
{1,2,10,9,5,11,9,11,8,11,5,6,-1,-1,-1,-1},
{0,11,3,0,6,11,0,9,6,5,6,9,1,2,10,-1},
{11,8,5,11,5,6,8,0,5,10,5,2,0,2,5,-1},
{6,11,3,6,3,5,2,10,3,10,5,3,-1,-1,-1,-1},
{5,8,9,5,2,8,5,6,2,3,8,2,-1,-1,-1,-1},
{9,5,6,9,6,0,0,6,2,-1,-1,-1,-1,-1,-1,-1},
{1,5,8,1,8,0,5,6,8,3,8,2,6,2,8,-1},
{1,5,6,2,1,6,-1,-1,-1,-1,-1,-1,-1,-1,-1,-1},
{1,3,6,1,6,10,3,8,6,5,6,9,8,9,6,-1},
{10,1,0,10,0,6,9,5,0,5,6,0,-1,-1,-1,-1},
{0,3,8,5,6,10,-1,-1,-1,-1,-1,-1,-1,-1,-1,-1},
{10,5,6,-1,-1,-1,-1,-1,-1,-1,-1,-1,-1,-1,-1,-1},
{11,5,10,7,5,11,-1,-1,-1,-1,-1,-1,-1,-1,-1,-1},
{11,5,10,11,7,5,8,3,0,-1,-1,-1,-1,-1,-1,-1},
{5,11,7,5,10,11,1,9,0,-1,-1,-1,-1,-1,-1,-1},
{10,7,5,10,11,7,9,8,1,8,3,1,-1,-1,-1,-1},
{11,1,2,11,7,1,7,5,1,-1,-1,-1,-1,-1,-1,-1},
{0,8,3,1,2,7,1,7,5,7,2,11,-1,-1,-1,-1},
{9,7,5,9,2,7,9,0,2,2,11,7,-1,-1,-1,-1},
{7,5,2,7,2,11,5,9,2,3,2,8,9,8,2,-1},
{2,5,10,2,3,5,3,7,5,-1,-1,-1,-1,-1,-1,-1},
{8,2,0,8,5,2,8,7,5,10,2,5,-1,-1,-1,-1},
{9,0,1,5,10,3,5,3,7,3,10,2,-1,-1,-1,-1},
{9,8,2,9,2,1,8,7,2,10,2,5,7,5,2,-1},
{1,3,5,3,7,5,-1,-1,-1,-1,-1,-1,-1,-1,-1,-1},
{0,8,7,0,7,1,1,7,5,-1,-1,-1,-1,-1,-1,-1},
{9,0,3,9,3,5,5,3,7,-1,-1,-1,-1,-1,-1,-1},
{9,8,7,5,9,7,-1,-1,-1,-1,-1,-1,-1,-1,-1,-1},
{5,8,4,5,10,8,10,11,8,-1,-1,-1,-1,-1,-1,-1},
{5,0,4,5,11,0,5,10,11,11,3,0,-1,-1,-1,-1},
{0,1,9,8,4,10,8,10,11,10,4,5,-1,-1,-1,-1},
{10,11,4,10,4,5,11,3,4,9,4,1,3,1,4,-1},
{2,5,1,2,8,5,2,11,8,4,5,8,-1,-1,-1,-1},
{0,4,11,0,11,3,4,5,11,2,11,1,5,1,11,-1},
{0,2,5,0,5,9,2,11,5,4,5,8,11,8,5,-1},
{9,4,5,2,11,3,-1,-1,-1,-1,-1,-1,-1,-1,-1,-1},
{2,5,10,3,5,2,3,4,5,3,8,4,-1,-1,-1,-1},
{5,10,2,5,2,4,4,2,0,-1,-1,-1,-1,-1,-1,-1},
{3,10,2,3,5,10,3,8,5,4,5,8,0,1,9,-1},
{5,10,2,5,2,4,1,9,2,9,4,2,-1,-1,-1,-1},
{8,4,5,8,5,3,3,5,1,-1,-1,-1,-1,-1,-1,-1},
{0,4,5,1,0,5,-1,-1,-1,-1,-1,-1,-1,-1,-1,-1},
{8,4,5,8,5,3,9,0,5,0,3,5,-1,-1,-1,-1},
{9,4,5,-1,-1,-1,-1,-1,-1,-1,-1,-1,-1,-1,-1,-1},
{4,11,7,4,9,11,9,10,11,-1,-1,-1,-1,-1,-1,-1},
{0,8,3,4,9,7,9,11,7,9,10,11,-1,-1,-1,-1},
{1,10,11,1,11,4,1,4,0,7,4,11,-1,-1,-1,-1},
{3,1,4,3,4,8,1,10,4,7,4,11,10,11,4,-1},
{4,11,7,9,11,4,9,2,11,9,1,2,-1,-1,-1,-1},
{9,7,4,9,11,7,9,1,11,2,11,1,0,8,3,-1},
{11,7,4,11,4,2,2,4,0,-1,-1,-1,-1,-1,-1,-1},
{11,7,4,11,4,2,8,3,4,3,2,4,-1,-1,-1,-1},
{2,9,10,2,7,9,2,3,7,7,4,9,-1,-1,-1,-1},
{9,10,7,9,7,4,10,2,7,8,7,0,2,0,7,-1},
{3,7,10,3,10,2,7,4,10,1,10,0,4,0,10,-1},
{1,10,2,8,7,4,-1,-1,-1,-1,-1,-1,-1,-1,-1,-1},
{4,9,1,4,1,7,7,1,3,-1,-1,-1,-1,-1,-1,-1},
{4,9,1,4,1,7,0,8,1,8,7,1,-1,-1,-1,-1},
{4,0,3,7,4,3,-1,-1,-1,-1,-1,-1,-1,-1,-1,-1},
{4,8,7,-1,-1,-1,-1,-1,-1,-1,-1,-1,-1,-1,-1,-1},
{9,10,8,10,11,8,-1,-1,-1,-1,-1,-1,-1,-1,-1,-1},
{3,0,9,3,9,11,11,9,10,-1,-1,-1,-1,-1,-1,-1},
{0,1,10,0,10,8,8,10,11,-1,-1,-1,-1,-1,-1,-1},
{3,1,10,11,3,10,-1,-1,-1,-1,-1,-1,-1,-1,-1,-1},
{1,2,11,1,11,9,9,11,8,-1,-1,-1,-1,-1,-1,-1},
{3,0,9,3,9,11,1,2,9,2,11,9,-1,-1,-1,-1},
{0,2,11,8,0,11,-1,-1,-1,-1,-1,-1,-1,-1,-1,-1},
{3,2,11,-1,-1,-1,-1,-1,-1,-1,-1,-1,-1,-1,-1,-1},
{2,3,8,2,8,10,10,8,9,-1,-1,-1,-1,-1,-1,-1},
{9,10,2,0,9,2,-1,-1,-1,-1,-1,-1,-1,-1,-1,-1},
{2,3,8,2,8,10,0,1,8,1,10,8,-1,-1,-1,-1},
{1,10,2,-1,-1,-1,-1,-1,-1,-1,-1,-1,-1,-1,-1,-1},
{1,3,8,9,1,8,-1,-1,-1,-1,-1,-1,-1,-1,-1,-1},
{0,9,1,-1,-1,-1,-1,-1,-1,-1,-1,-1,-1,-1,-1,-1},
{0,3,8,-1,-1,-1,-1,-1,-1,-1,-1,-1,-1,-1,-1,-1},
{-1,-1,-1,-1,-1,-1,-1,-1,-1,-1,-1,-1,-1,-1,-1,-1}};

__constant__ unsigned char c_ea[12] = {0,1,2,3,4,5,6,7,0,1,2,3};
__constant__ unsigned char c_eb[12] = {1,2,3,0,5,6,7,4,4,5,6,7};
__constant__ float c_cx[8] = {0.f,1.f,1.f,0.f,0.f,1.f,1.f,0.f};
__constant__ float c_cy[8] = {0.f,0.f,1.f,1.f,0.f,0.f,1.f,1.f};
__constant__ float c_cz[8] = {0.f,0.f,0.f,0.f,1.f,1.f,1.f,1.f};
__constant__ int   c_ox[8] = {0,1,1,0,0,1,1,0};
__constant__ int   c_oy[8] = {0,0,1,1,0,0,1,1};
__constant__ int   c_oz[8] = {0,0,0,0,1,1,1,1};

__global__ void mc_zero(float* p) { *p = 0.0f; }

// One block = one (y,z) cell row; 128 threads along x (thread 127 pads to full waves).
// TDM DMAs the 128x2x2 corner slab into LDS; edge vertices staged in LDS for the
// table-indexed gather; WMMA does the cross-lane pair-sum for triangle counting.
__global__ __launch_bounds__(128) void mc_kernel(const float* __restrict__ vol,
                                                 float* __restrict__ out)
{
    __shared__ float tile[4 * VDIM];       // [(zz*2+yy)*128 + x]
    __shared__ float s_ev[128 * 37];       // per-thread 36 edge-vertex floats, stride 37 (bank-clean)
    __shared__ float s_cnt;

    const int x = threadIdx.x;
    const int y = blockIdx.x;
    const int z = blockIdx.y;

    if (x == 0) s_cnt = 0.0f;

#if __has_builtin(__builtin_amdgcn_tensor_load_to_lds)
    if (x < 32) {  // wave 0 only: TDM issues regardless of EXEC, so branch-guard it
        unsigned long long ga = (unsigned long long)(uintptr_t)vol
            + 4ull * ((unsigned long long)z * VDIM * VDIM + (unsigned long long)y * VDIM);
        unsigned int lds_off = (unsigned int)(uintptr_t)(&tile[0]); // low 32 bits of generic ptr = LDS byte offset
        u32x4 g0;
        g0[0] = 1u;                                   // count=1 (valid descriptor), gather off
        g0[1] = lds_off;                              // lds_addr
        g0[2] = (unsigned int)ga;                     // global_addr[31:0]
        g0[3] = (unsigned int)(ga >> 32) | (2u << 30);// global_addr[56:32] | type=2 ("image")
        i32x8 g1;
        g1[0] = (int)(2u << 16);                      // workgroup_mask=0, data_size=2 (4B)
        g1[1] = (int)(((unsigned)VDIM) << 16);        // tensor_dim0 = 128 (lo16)
        g1[2] = (int)(((unsigned)VDIM) << 16);        // tensor_dim0 hi=0 | tensor_dim1=128 (lo16)
        g1[3] = (int)(((unsigned)VDIM) << 16);        // tensor_dim1 hi=0 | tile_dim0=128
        g1[4] = (int)(2u | (2u << 16));               // tile_dim1=2, tile_dim2=2
        g1[5] = VDIM;                                 // tensor_dim0_stride = 128 elements
        g1[6] = (int)(((unsigned)(VDIM * VDIM)) << 16); // stride0 hi=0 | tensor_dim1_stride lo16 = 16384
        g1[7] = 0;                                    // tensor_dim1_stride hi
        i32x4 g2; g2[0] = VDIM; g2[1] = 0; g2[2] = 0; g2[3] = 0;  // tensor_dim2=128, 3D tile
        i32x4 g3; g3[0] = 0; g3[1] = 0; g3[2] = 0; g3[3] = 0;
        i32x8 g4; g4[0]=0; g4[1]=0; g4[2]=0; g4[3]=0; g4[4]=0; g4[5]=0; g4[6]=0; g4[7]=0;
        // clang-23 toolchain: 6-arg form (g0, g1, g2, g3, <i32x8 extra, zero-filled>, cpol)
        __builtin_amdgcn_tensor_load_to_lds(g0, g1, g2, g3, g4, 0);
        __builtin_amdgcn_s_wait_tensorcnt(0);
    }
#else
    for (int i = x; i < 4 * VDIM; i += 128) {
        int xx = i & 127, yy = (i >> 7) & 1, zz = i >> 8;
        tile[i] = vol[((size_t)(z + zz) * VDIM + (size_t)(y + yy)) * VDIM + xx];
    }
#endif
    __syncthreads();

    float cnt = 0.0f;
    if (x < CDIM) {
        float cv[8];
        int cube = 0;
        #pragma unroll
        for (int c = 0; c < 8; ++c) {
            cv[c] = tile[(c_oz[c] * 2 + c_oy[c]) * VDIM + x + c_ox[c]];
            cube |= (cv[c] < 0.5f) ? (1 << c) : 0;
        }

        const float bx = (float)x, by = (float)y, bz = (float)z;
        float* ev = &s_ev[x * 37];
        #pragma unroll
        for (int e = 0; e < 12; ++e) {
            const int a = c_ea[e], b = c_eb[e];
            const float v1 = cv[a];
            const float den = cv[b] - v1;
            const float t = (__builtin_fabsf(den) > 1e-9f) ? (0.5f - v1) / den : 0.5f;
            ev[e * 3 + 0] = bx + c_cx[a] + t * (c_cx[b] - c_cx[a]);
            ev[e * 3 + 1] = by + c_cy[a] + t * (c_cy[b] - c_cy[a]);
            ev[e * 3 + 2] = bz + c_cz[a] + t * (c_cz[b] - c_cz[a]);
        }

        int4 raw = *(const int4*)c_tri[cube];         // one 16B load of the row
        const signed char* tr = (const signed char*)&raw;

        const size_t cell = ((size_t)z * CDIM + (size_t)y) * CDIM + (size_t)x;
        float* __restrict__ vtx = out + cell * 45ull;
        float* __restrict__ tvd = out + NCELL * 45ull + cell * 5ull;

        #pragma unroll
        for (int t5 = 0; t5 < 5; ++t5) {
            const int e0 = (int)tr[t5 * 3];
            const float valid = (e0 >= 0) ? 1.0f : 0.0f;
            cnt += valid;
            #pragma unroll
            for (int v = 0; v < 3; ++v) {
                int idx = (int)tr[t5 * 3 + v];
                idx = idx < 0 ? 0 : idx;
                vtx[t5 * 9 + v * 3 + 0] = ev[idx * 3 + 0] * valid;
                vtx[t5 * 9 + v * 3 + 1] = ev[idx * 3 + 1] * valid;
                vtx[t5 * 9 + v * 3 + 2] = ev[idx * 3 + 2] * valid;
            }
            tvd[t5] = valid;
        }
    }

    // Cross-lane pair reduction via WMMA: A = ones(16x32 f16), lane L puts its count
    // (<=5, exact in f16) in B element 0 -> D[m,n] = c_n + c_{n+16} for every m.
    v16h av, bv;
    #pragma unroll
    for (int i = 0; i < 16; ++i) { av[i] = (_Float16)1.0f; bv[i] = (_Float16)0.0f; }
    bv[0] = (_Float16)cnt;
    v8f acc = {};
    acc = __builtin_amdgcn_wmma_f32_16x16x32_f16(false, av, false, bv,
                                                 (short)0, acc, false, false);
    if ((threadIdx.x & 31) < 16) atomicAdd(&s_cnt, acc[0]);  // 16 pair-sums per wave
    __syncthreads();
    if (threadIdx.x == 0) atomicAdd(out + NCELL * 50ull, s_cnt);
}

extern "C" void kernel_launch(void* const* d_in, const int* in_sizes, int n_in,
                              void* d_out, int out_size, void* d_ws, size_t ws_size,
                              hipStream_t stream)
{
    (void)in_sizes; (void)n_in; (void)out_size; (void)d_ws; (void)ws_size;
    const float* vol = (const float*)d_in[0];
    float* out = (float*)d_out;
    mc_zero<<<1, 1, 0, stream>>>(out + NCELL * 50ull);
    dim3 grid(CDIM, CDIM, 1);                // blockIdx.x = y-cell, blockIdx.y = z-cell
    mc_kernel<<<grid, dim3(128, 1, 1), 0, stream>>>(vol, out);
}
